// TFTModel_69569880260657
// MI455X (gfx1250) — matrix operand order repaired and checked by reading, not compiled, not run on hardware
//
#include <hip/hip_runtime.h>

typedef __attribute__((ext_vector_type(16))) __bf16 v16bf;
typedef __attribute__((ext_vector_type(8)))  float  v8f;
typedef unsigned short u16;

#define DEV static __device__ __forceinline__

// native f32 -> bf16 (RNE) conversion; lowers to HW cvt on gfx1250
DEV u16 f2bf(float x){ return __builtin_bit_cast(u16, (__bf16)x); }
DEV float bf2f(u16 h){ unsigned u = ((unsigned)h) << 16; return __builtin_bit_cast(float, u); }

// store two adjacent bf16 halves as one 32-bit DS store
DEV void st_pair(u16* p, float v0, float v1){
  unsigned pk = (unsigned)f2bf(v0) | ((unsigned)f2bf(v1) << 16);
  *(unsigned*)p = pk;
}

// Index of element (m,k) in a packed WMMA-A buffer (16x32 bf16 tiles, contiguous k = 16*(lane/16) + s,
// multiple k-tiles contiguous, 512 halves per tile).
DEV int a_idx(int m, int k){
  int kt = k >> 5, k32 = k & 31;
  int lane = m + ((k32 >> 4) << 4);
  int h = k32 & 15;
  return kt * 512 + lane * 16 + h;
}
// Index of element (k,n) within one packed WMMA-B tile (32x16 bf16, 512 halves).
DEV int b_idx(int k, int n){
  return ((n & 15) + ((k >> 4) << 4)) * 16 + (k & 15);
}
DEV v16bf ld16(const u16* p){ return *(const v16bf*)p; }

DEV v8f wmma_bf16(v16bf a, v16bf b, v8f c){
  return __builtin_amdgcn_wmma_f32_16x16x32_bf16(false, a, false, b, (short)0, c, false, false);
}
DEV v8f zero8(){ v8f z = {0.f,0.f,0.f,0.f,0.f,0.f,0.f,0.f}; return z; }
DEV float eluf(float x){ return x > 0.f ? x : __expf(x) - 1.f; }
DEV float sigm(float x){ return 1.f / (1.f + __expf(-x)); }
DEV void wbar(){ __builtin_amdgcn_wave_barrier(); }

// CDNA5 async DMA: global -> LDS, 16B per lane, tracked by ASYNCcnt.
DEV void async_b128(unsigned lds_off, const void* gaddr){
  asm volatile("global_load_async_to_lds_b128 %0, %1, off"
               :: "v"(lds_off), "v"(gaddr) : "memory");
}
DEV void wait_async0(){ asm volatile("s_wait_asynccnt 0x0" ::: "memory"); }

// contiguous packed-weight sub-block offsets (in halves; host layout guarantees)
constexpr int FG1_OFF = 331776;   // f_g1w tiles after f_w2 tiles
constexpr int FG2_OFF = 663552;   // f_g2w tiles after f_g1w tiles
constexpr int WSK_OFF = 331776;   // w_skw tiles after w_w1 tiles
constexpr int WW1_NTSTRIDE = 162 * 512;  // halves per n-tile column of w_w1/w_skw

// ---------------------------------------------------------------------------
// Pack f32 matrix (element (k,n) at src[k*strideK + n*strideN]) into bf16
// WMMA-B tiles. Tile order: tile = nt*KT + kt. Pads with zeros beyond K,N.
// ---------------------------------------------------------------------------
__global__ void pack_b(const float* __restrict__ src, u16* __restrict__ dst,
                       int K, int N, int KT, int NT, int strideK, int strideN,
                       long srcBS, long dstBS){
  int b = blockIdx.y;
  long id = (long)blockIdx.x * blockDim.x + threadIdx.x;
  long nHalf = (long)KT * NT * 512;
  if(id >= nHalf) return;
  int tile = (int)(id >> 9);
  int within = (int)(id & 511);
  int kt = tile % KT, nt = tile / KT;
  int lane = within >> 4, h = within & 15;
  int k = kt * 32 + ((lane >> 4) << 4) + h;
  int n = nt * 16 + (lane & 15);
  float v = 0.f;
  if(k < K && n < N) v = src[(long)b * srcBS + (long)k * strideK + (long)n * strideN];
  dst[(long)b * dstBS + id] = f2bf(v);
}

// ---------------------------------------------------------------------------
// K1: fused Variable Selection Network.
// 768 workgroups x 128 threads (4 waves). Each WG owns 16 rows of [B*T].
// Phase 1 (per wave): 81 per-feature GRNs -> `stacked` resident in LDS in
// WMMA-A-packed bf16 layout; all three weight matrices addressed from ONE
// per-feature base pointer with constant immediate offsets (no addr spills).
// Phase 2: accumulate flat@w_w1 / flat@w_skw. Then tail + softmax + mix.
// ---------------------------------------------------------------------------
constexpr int K1_XT   = 165888;            // stacked: u16[81][1024] at 0 (A-packed)
constexpr int K1_ABUF = K1_XT + 5184;      // xt: float[16*81]
constexpr int K1_HBUF = K1_ABUF + 8192;    // abufA: u16[4][1024]
constexpr int K1_SCR  = K1_HBUF + 8192;    // abufH: u16[4][1024]
constexpr int K1_LNST = K1_SCR + 16384;    // scr: float[4][1024]
constexpr int K1_WHR  = K1_LNST + 512;     // lnst: float[4][32]
constexpr int K1_SKR  = K1_WHR + 16384;    // whr: float[4][1024]
constexpr int K1_TA2  = K1_SKR + 24576;    // skr: float[4][1536]
constexpr int K1_TS   = K1_TA2 + 3072;     // tA2: u16[1536]
constexpr int K1_WSM  = K1_TS + 6144;      // tS: float[16*96]
constexpr int K1_LDS  = K1_WSM + 6144;     // wsm: float[16*96]  -> total 260672 B

__global__ void k1_vsn(
  const float* __restrict__ x,
  const float* __restrict__ f_w1, const float* __restrict__ f_b1,
  const float* __restrict__ f_b2, const float* __restrict__ f_g1b,
  const float* __restrict__ f_g2b, const float* __restrict__ f_skw,
  const float* __restrict__ f_skb, const float* __restrict__ f_lng,
  const float* __restrict__ f_lnb,
  const u16* __restrict__ pw_fw2,            // FW2 | FG1 | FG2 contiguous
  const u16* __restrict__ pw_ww1,            // WW1 | WSK contiguous
  const float* __restrict__ w_b1, const u16* __restrict__ pw_ww2,
  const float* __restrict__ w_b2, const u16* __restrict__ pw_wg1,
  const float* __restrict__ w_g1b, const u16* __restrict__ pw_wg2,
  const float* __restrict__ w_g2b, const float* __restrict__ w_skb,
  const float* __restrict__ w_lng, const float* __restrict__ w_lnb,
  float* __restrict__ selected)
{
  extern __shared__ char smem[];
  u16* stacked = (u16*)smem;                 // [81][1024] A-packed
  float* xt = (float*)(smem + K1_XT);
  int tid = threadIdx.x, wave = tid >> 5, lane = tid & 31;
  u16* abufA = (u16*)(smem + K1_ABUF) + wave * 1024;
  u16* abufH = (u16*)(smem + K1_HBUF) + wave * 1024;
  float* scr  = (float*)(smem + K1_SCR) + wave * 1024;
  float* lnst = (float*)(smem + K1_LNST) + wave * 32;
  float* whr  = (float*)(smem + K1_WHR);
  float* skr  = (float*)(smem + K1_SKR);
  u16* tA2    = (u16*)(smem + K1_TA2);
  float* tS   = (float*)(smem + K1_TS);
  float* wsm  = (float*)(smem + K1_WSM);
  int cr0 = ((lane >> 4) << 3), cc = lane & 15;
  long row0 = (long)blockIdx.x * 16;

  for(int i = tid; i < 16 * 81; i += 128){
    int m = i / 81, f = i % 81;
    xt[m * 81 + f] = x[(row0 + m) * 81 + f];
  }
  __syncthreads();

  // ---------------- Phase 1: per-feature GRNs -> stacked (packed) ----------
  for(int f = wave; f < 81; f += 4){
    const float* w1  = f_w1 + f * 64;  const float* b1  = f_b1 + f * 64;
    const float* b2  = f_b2 + f * 64;  const float* g1b = f_g1b + f * 64;
    const float* g2b = f_g2b + f * 64; const float* skw = f_skw + f * 64;
    const float* skb = f_skb + f * 64; const float* lng = f_lng + f * 64;
    const float* lnb = f_lnb + f * 64;
    const u16* base = pw_fw2 + (long)f * 4096 + lane * 16;  // single addr stream
    if(f + 4 < 81){
      __builtin_prefetch(pw_fw2 + (long)(f + 4) * 4096, 0, 0);
      __builtin_prefetch(pw_fw2 + FG1_OFF + (long)(f + 4) * 4096, 0, 0);
      __builtin_prefetch(pw_fw2 + FG2_OFF + (long)(f + 4) * 4096, 0, 0);
    }
    // hoist per-feature bias/skip scalars so their latency overlaps compute
    float bb2[4], bg1[4], bg2[4], bsw[4], bsb[4];
    #pragma unroll
    for(int nt = 0; nt < 4; nt++){
      int n = nt * 16 + cc;
      bb2[nt] = b2[n]; bg1[nt] = g1b[n]; bg2[nt] = g2b[n];
      bsw[nt] = skw[n]; bsb[nt] = skb[n];
    }
    // h = elu(x*w1 + b1) -> packed A (paired b32 stores)
    for(int i = lane; i < 512; i += 32){
      int m = i >> 5, d = (i & 31) * 2;
      float xv = xt[m * 81 + f];
      st_pair(abufA + a_idx(m, d),
              eluf(xv * w1[d] + b1[d]),
              eluf(xv * w1[d + 1] + b1[d + 1]));
    }
    wbar();
    v16bf a0 = ld16(abufA + lane * 16), a1 = ld16(abufA + 512 + lane * 16);
    // h2 = h @ f_w2 + b2 : preload all 8 B tiles, 4 independent accumulators
    v16bf bw[8];
    #pragma unroll
    for(int j = 0; j < 8; j++) bw[j] = ld16(base + j * 512);
    v8f h2a[4];
    #pragma unroll
    for(int nt = 0; nt < 4; nt++){
      h2a[nt] = wmma_bf16(a0, bw[2 * nt + 0], zero8());
      h2a[nt] = wmma_bf16(a1, bw[2 * nt + 1], h2a[nt]);
    }
    #pragma unroll
    for(int nt = 0; nt < 4; nt++)
      #pragma unroll
      for(int r = 0; r < 8; r++)
        abufH[a_idx(cr0 + r, nt * 16 + cc)] = f2bf(h2a[nt][r] + bb2[nt]);
    wbar();
    v16bf h0 = ld16(abufH + lane * 16), h1 = ld16(abufH + 512 + lane * 16);
    // GLU gates: preload G1 tiles -> c1a[4], then G2 tiles -> c2a[4]
    #pragma unroll
    for(int j = 0; j < 8; j++) bw[j] = ld16(base + FG1_OFF + j * 512);
    v8f c1a[4];
    #pragma unroll
    for(int nt = 0; nt < 4; nt++){
      c1a[nt] = wmma_bf16(h0, bw[2 * nt + 0], zero8());
      c1a[nt] = wmma_bf16(h1, bw[2 * nt + 1], c1a[nt]);
    }
    #pragma unroll
    for(int j = 0; j < 8; j++) bw[j] = ld16(base + FG2_OFF + j * 512);
    v8f c2a[4];
    #pragma unroll
    for(int nt = 0; nt < 4; nt++){
      c2a[nt] = wmma_bf16(h0, bw[2 * nt + 0], zero8());
      c2a[nt] = wmma_bf16(h1, bw[2 * nt + 1], c2a[nt]);
    }
    #pragma unroll
    for(int nt = 0; nt < 4; nt++){
      int n = nt * 16 + cc;
      #pragma unroll
      for(int r = 0; r < 8; r++){
        int m = cr0 + r;
        scr[m * 64 + n] = sigm(c1a[nt][r] + bg1[nt]) * (c2a[nt][r] + bg2[nt])
                        + xt[m * 81 + f] * bsw[nt] + bsb[nt];
      }
    }
    wbar();
    // LayerNorm over D=64
    if(lane < 16){
      float s = 0.f, ss = 0.f;
      for(int d = 0; d < 64; d++){ float v = scr[lane * 64 + d]; s += v; ss += v * v; }
      float mean = s * (1.f / 64.f);
      lnst[lane * 2] = mean;
      lnst[lane * 2 + 1] = rsqrtf(ss * (1.f / 64.f) - mean * mean + 1e-5f);
    }
    wbar();
    u16* sp = stacked + f * 1024;
    for(int i = lane; i < 512; i += 32){
      int m = i >> 5, d = (i & 31) * 2;
      float mu = lnst[m * 2], rs = lnst[m * 2 + 1];
      float v0 = (scr[m * 64 + d]     - mu) * rs * lng[d]     + lnb[d];
      float v1 = (scr[m * 64 + d + 1] - mu) * rs * lng[d + 1] + lnb[d + 1];
      st_pair(sp + a_idx(m, d), v0, v1);
    }
    wbar();
  }

  // ---------------- Phase 2: weight-GRN projection accumulation -----------
  v8f wh[4], sk[6];
  #pragma unroll
  for(int i = 0; i < 4; i++) wh[i] = zero8();
  #pragma unroll
  for(int i = 0; i < 6; i++) sk[i] = zero8();
  for(int f = wave; f < 81; f += 4){
    const u16* sp = stacked + f * 1024 + lane * 16;
    v16bf s0 = ld16(sp), s1 = ld16(sp + 512);
    const u16* wb = pw_ww1 + (long)f * 1024 + lane * 16;  // single addr stream
    if(f + 4 < 81){
      __builtin_prefetch(pw_ww1 + (long)(f + 4) * 1024, 0, 0);
      __builtin_prefetch(pw_ww1 + WSK_OFF + (long)(f + 4) * 1024, 0, 0);
    }
    #pragma unroll
    for(int nt = 0; nt < 4; nt++){
      wh[nt] = wmma_bf16(s0, ld16(wb + nt * WW1_NTSTRIDE + 0), wh[nt]);
      wh[nt] = wmma_bf16(s1, ld16(wb + nt * WW1_NTSTRIDE + 512), wh[nt]);
    }
    #pragma unroll
    for(int nt = 0; nt < 6; nt++){
      sk[nt] = wmma_bf16(s0, ld16(wb + WSK_OFF + nt * WW1_NTSTRIDE + 0), sk[nt]);
      sk[nt] = wmma_bf16(s1, ld16(wb + WSK_OFF + nt * WW1_NTSTRIDE + 512), sk[nt]);
    }
  }
  // cross-wave reduction of accumulators
  #pragma unroll
  for(int nt = 0; nt < 4; nt++)
    #pragma unroll
    for(int r = 0; r < 8; r++)
      whr[wave * 1024 + (cr0 + r) * 64 + nt * 16 + cc] = wh[nt][r];
  #pragma unroll
  for(int nt = 0; nt < 6; nt++)
    #pragma unroll
    for(int r = 0; r < 8; r++)
      skr[wave * 1536 + (cr0 + r) * 96 + nt * 16 + cc] = sk[nt][r];
  __syncthreads();
  for(int i = tid; i < 1024; i += 128) whr[i] += whr[1024 + i] + whr[2048 + i] + whr[3072 + i];
  for(int i = tid; i < 1536; i += 128) skr[i] += skr[1536 + i] + skr[3072 + i] + skr[4608 + i];
  __syncthreads();

  if(wave == 0){ // weight-GRN tail + softmax (wave 0 only)
    u16* tA = abufA;
    for(int i = lane; i < 512; i += 32){
      int m = i >> 5, d = (i & 31) * 2;
      st_pair(tA + a_idx(m, d),
              eluf(whr[m * 64 + d] + w_b1[d]),
              eluf(whr[m * 64 + d + 1] + w_b1[d + 1]));
    }
    wbar();
    v16bf t0 = ld16(tA + lane * 16), t1 = ld16(tA + 512 + lane * 16);
    #pragma unroll
    for(int nt = 0; nt < 6; nt++){
      v8f acc = zero8();
      acc = wmma_bf16(t0, ld16(pw_ww2 + (nt * 2 + 0) * 512 + lane * 16), acc);
      acc = wmma_bf16(t1, ld16(pw_ww2 + (nt * 2 + 1) * 512 + lane * 16), acc);
      int n = nt * 16 + cc;
      float bb = (n < 81) ? w_b2[n] : 0.f;
      #pragma unroll
      for(int r = 0; r < 8; r++)
        tA2[a_idx(cr0 + r, n)] = f2bf((n < 81) ? acc[r] + bb : 0.f);
    }
    wbar();
    v16bf q0 = ld16(tA2 + lane * 16), q1 = ld16(tA2 + 512 + lane * 16), q2 = ld16(tA2 + 1024 + lane * 16);
    #pragma unroll
    for(int nt = 0; nt < 6; nt++){
      v8f c1 = zero8(), c2 = zero8();
      c1 = wmma_bf16(q0, ld16(pw_wg1 + (nt * 3 + 0) * 512 + lane * 16), c1);
      c1 = wmma_bf16(q1, ld16(pw_wg1 + (nt * 3 + 1) * 512 + lane * 16), c1);
      c1 = wmma_bf16(q2, ld16(pw_wg1 + (nt * 3 + 2) * 512 + lane * 16), c1);
      c2 = wmma_bf16(q0, ld16(pw_wg2 + (nt * 3 + 0) * 512 + lane * 16), c2);
      c2 = wmma_bf16(q1, ld16(pw_wg2 + (nt * 3 + 1) * 512 + lane * 16), c2);
      c2 = wmma_bf16(q2, ld16(pw_wg2 + (nt * 3 + 2) * 512 + lane * 16), c2);
      int n = nt * 16 + cc;
      float g1v = (n < 81) ? w_g1b[n] : 0.f;
      float g2v = (n < 81) ? w_g2b[n] : 0.f;
      float sbv = (n < 81) ? w_skb[n] : 0.f;
      #pragma unroll
      for(int r = 0; r < 8; r++){
        int m = cr0 + r;
        tS[m * 96 + n] = (n < 81) ? sigm(c1[r] + g1v) * (c2[r] + g2v) + skr[m * 96 + n] + sbv : 0.f;
      }
    }
    wbar();
    if(lane < 16){
      int m = lane;
      float s = 0.f, ss = 0.f;
      for(int n = 0; n < 81; n++){ float v = tS[m * 96 + n]; s += v; ss += v * v; }
      float mean = s * (1.f / 81.f);
      float rstd = rsqrtf(ss * (1.f / 81.f) - mean * mean + 1e-5f);
      float mx = -1e30f;
      for(int n = 0; n < 81; n++){
        float v = (tS[m * 96 + n] - mean) * rstd * w_lng[n] + w_lnb[n];
        wsm[m * 96 + n] = v;
        mx = v > mx ? v : mx;
      }
      float es = 0.f;
      for(int n = 0; n < 81; n++){ float e = __expf(wsm[m * 96 + n] - mx); wsm[m * 96 + n] = e; es += e; }
      float inv = 1.f / es;
      for(int n = 0; n < 81; n++) wsm[m * 96 + n] *= inv;
    }
  }
  __syncthreads();
  // selected = sum_f stacked[:,f,:] * weights[:,f]  (stacked is A-packed)
  for(int i = tid; i < 1024; i += 128){
    int m = i >> 6, d = i & 63;
    int base2 = a_idx(m, d);
    float acc = 0.f;
    for(int f = 0; f < 81; f++)
      acc += bf2f(stacked[f * 1024 + base2]) * wsm[m * 96 + f];
    selected[(row0 + m) * 64 + d] = acc;
  }
}

// ---------------------------------------------------------------------------
// K2: LSTM. 32 workgroups x 128 threads; each WG owns 16 batch rows and loops
// t = 0..23 with h (packed bf16 A) and c resident in LDS. The 64 KB of packed
// wih^T/whh^T tiles are staged into LDS once per WG via CDNA5 async DMA
// (GLOBAL_LOAD_ASYNC_TO_LDS_B128, ASYNCcnt) and reused for all 24 steps.
// ---------------------------------------------------------------------------
__global__ void k2_lstm(const float* __restrict__ sel,
                        const u16* __restrict__ pw_wih, const u16* __restrict__ pw_whh,
                        const float* __restrict__ bih, const float* __restrict__ bhh,
                        float* __restrict__ lout){
  extern __shared__ char smem[];
  u16* hA = (u16*)smem;                 // 1024 halves
  u16* xA = hA + 1024;                  // 1024 halves
  float* gates = (float*)(smem + 4096); // [4][1024]
  float* cst = (float*)(smem + 20480);  // [1024]
  u16* wst = (u16*)(smem + 24576);      // 64KB: wih tiles @0, whh tiles @32768B
  int tid = threadIdx.x, wave = tid >> 5, lane = tid & 31;
  int cr0 = ((lane >> 4) << 3), cc = lane & 15;
  long b0 = (long)blockIdx.x * 16;

  { // async-stage both weight matrices (each wave copies an 8KB slice of each)
    unsigned base = (unsigned)(size_t)wst;
    #pragma unroll
    for(int blk = 0; blk < 16; blk++){
      unsigned off = (unsigned)(wave * 8192 + blk * 512 + lane * 16);
      async_b128(base + off,          (const char*)pw_wih + off);
      async_b128(base + 32768u + off, (const char*)pw_whh + off);
    }
  }
  for(int i = tid; i < 1024; i += 128){ hA[i] = 0; cst[i] = 0.f; }
  wait_async0();
  __syncthreads();
  const u16* Wih = wst;                 // LDS-resident packed B tiles
  const u16* Whh = wst + 16384;         // halves offset (32768 bytes)

  for(int t = 0; t < 24; t++){
    for(int i = tid; i < 512; i += 128){
      int m = i >> 5, d = (i & 31) * 2;
      const float* spp = sel + ((b0 + m) * 24 + t) * 64 + d;
      st_pair(xA + a_idx(m, d), spp[0], spp[1]);
    }
    __syncthreads();
    v16bf x0 = ld16(xA + lane * 16), x1 = ld16(xA + 512 + lane * 16);
    v16bf h0 = ld16(hA + lane * 16), h1 = ld16(hA + 512 + lane * 16);
    #pragma unroll
    for(int nt = 0; nt < 4; nt++){
      int ng = wave * 4 + nt;
      v8f acc = zero8();
      acc = wmma_bf16(x0, ld16(Wih + (ng * 2 + 0) * 512 + lane * 16), acc);
      acc = wmma_bf16(x1, ld16(Wih + (ng * 2 + 1) * 512 + lane * 16), acc);
      acc = wmma_bf16(h0, ld16(Whh + (ng * 2 + 0) * 512 + lane * 16), acc);
      acc = wmma_bf16(h1, ld16(Whh + (ng * 2 + 1) * 512 + lane * 16), acc);
      int n = ng * 16 + cc;
      float bb = bih[n] + bhh[n];
      #pragma unroll
      for(int r = 0; r < 8; r++)
        gates[wave * 1024 + (cr0 + r) * 64 + nt * 16 + cc] = acc[r] + bb;
    }
    __syncthreads();
    for(int i = tid; i < 512; i += 128){
      int m = i >> 5, d = (i & 31) * 2;
      int i0 = m * 64 + d, i1 = i0 + 1;
      float c0 = sigm(gates[1024 + i0]) * cst[i0] + sigm(gates[i0]) * tanhf(gates[2048 + i0]);
      float h0v = sigm(gates[3072 + i0]) * tanhf(c0);
      float c1 = sigm(gates[1024 + i1]) * cst[i1] + sigm(gates[i1]) * tanhf(gates[2048 + i1]);
      float h1v = sigm(gates[3072 + i1]) * tanhf(c1);
      cst[i0] = c0; cst[i1] = c1;
      st_pair(hA + a_idx(m, d), h0v, h1v);
      lout[((b0 + m) * 24 + t) * 64 + d]     = h0v;
      lout[((b0 + m) * 24 + t) * 64 + d + 1] = h1v;
    }
    __syncthreads();
  }
}

// ---------------------------------------------------------------------------
// out = LN(sigmoid(in@g1+g1b) * (in@g2+g2b) + resid). 768 WGs x 32 threads.
// ---------------------------------------------------------------------------
__global__ void k_glu_ln(const float* __restrict__ inp, const float* __restrict__ resid,
                         const u16* __restrict__ pg1, const float* __restrict__ g1b,
                         const u16* __restrict__ pg2, const float* __restrict__ g2b,
                         const float* __restrict__ lng, const float* __restrict__ lnb,
                         float* __restrict__ outp){
  extern __shared__ char smem[];
  u16* ab = (u16*)smem;                // 1024 halves
  float* scr = (float*)(smem + 2048);  // 1024 floats
  float* st = (float*)(smem + 6144);   // 32 floats
  int lane = threadIdx.x;
  int cr0 = ((lane >> 4) << 3), cc = lane & 15;
  long r0 = (long)blockIdx.x * 16;
  for(int i = lane; i < 512; i += 32){
    int m = i >> 5, d = (i & 31) * 2;
    const float* ip = inp + (r0 + m) * 64 + d;
    st_pair(ab + a_idx(m, d), ip[0], ip[1]);
  }
  wbar();
  v16bf a0 = ld16(ab + lane * 16), a1 = ld16(ab + 512 + lane * 16);
  #pragma unroll
  for(int nt = 0; nt < 4; nt++){
    v8f c1 = zero8(), c2 = zero8();
    c1 = wmma_bf16(a0, ld16(pg1 + (nt * 2 + 0) * 512 + lane * 16), c1);
    c1 = wmma_bf16(a1, ld16(pg1 + (nt * 2 + 1) * 512 + lane * 16), c1);
    c2 = wmma_bf16(a0, ld16(pg2 + (nt * 2 + 0) * 512 + lane * 16), c2);
    c2 = wmma_bf16(a1, ld16(pg2 + (nt * 2 + 1) * 512 + lane * 16), c2);
    int n = nt * 16 + cc;
    float b1v = g1b[n], b2v = g2b[n];
    #pragma unroll
    for(int r = 0; r < 8; r++){
      int m = cr0 + r;
      scr[m * 64 + n] = sigm(c1[r] + b1v) * (c2[r] + b2v) + resid[(r0 + m) * 64 + n];
    }
  }
  wbar();
  if(lane < 16){
    float s = 0.f, ss = 0.f;
    for(int d = 0; d < 64; d++){ float v = scr[lane * 64 + d]; s += v; ss += v * v; }
    float mean = s * (1.f / 64.f);
    st[lane * 2] = mean;
    st[lane * 2 + 1] = rsqrtf(ss * (1.f / 64.f) - mean * mean + 1e-5f);
  }
  wbar();
  for(int i = lane; i < 1024; i += 32){
    int m = i >> 6, d = i & 63;
    outp[(r0 + m) * 64 + d] = (scr[m * 64 + d] - st[m * 2]) * st[m * 2 + 1] * lng[d] + lnb[d];
  }
}

// ---------------------------------------------------------------------------
// K4: multi-head self attention per batch element. 512 WGs x 128 threads;
// wave = head. T=24 rows padded to 32; head_dim 16 padded to K=32 for WMMA.
// ---------------------------------------------------------------------------
__global__ void k4_attn(const float* __restrict__ temporal,
                        const u16* __restrict__ pw_ain, const float* __restrict__ ain_b,
                        const u16* __restrict__ pw_aout, const float* __restrict__ aout_b,
                        float* __restrict__ attn_out){
  extern __shared__ char smem[];
  u16* tA   = (u16*)smem;              // [2][1024] temporal packed A
  u16* aQ   = (u16*)(smem + 4096);     // [4][1024]
  u16* bK   = (u16*)(smem + 12288);    // [4][1024]
  u16* bV   = (u16*)(smem + 20480);    // [4][512]
  u16* aAW  = (u16*)(smem + 24576);    // [4][1024]
  float* sb = (float*)(smem + 32768);  // [4][1024] scores
  u16* aAO  = (u16*)(smem + 49152);    // [2][1024] concat heads
  int tid = threadIdx.x, wave = tid >> 5, lane = tid & 31;
  int cr0 = ((lane >> 4) << 3), cc = lane & 15;
  long r0 = (long)blockIdx.x * 24;

  for(int i = tid; i < 1024; i += 128){
    int m = i >> 5, d = (i & 31) * 2;
    float v0 = 0.f, v1 = 0.f;
    if(m < 24){
      const float* tp = temporal + (r0 + m) * 64 + d;
      v0 = tp[0]; v1 = tp[1];
    }
    st_pair(tA + (m >> 4) * 1024 + a_idx(m & 15, d), v0, v1);
  }
  __syncthreads();

  u16* aQw = aQ + wave * 1024;
  u16* bKw = bK + wave * 1024;
  u16* bVw = bV + wave * 512;
  u16* aAWw = aAW + wave * 1024;
  float* sbw = sb + wave * 1024;
  for(int i = lane; i < 512; i += 32){
    aQw[i] = 0; aQw[512 + i] = 0;
    bKw[i] = 0; bKw[512 + i] = 0;
    bVw[i] = 0;
    aAWw[i] = 0; aAWw[512 + i] = 0;
  }
  wbar();

  int h = wave;
  #pragma unroll
  for(int mt = 0; mt < 2; mt++){
    v16bf a0 = ld16(tA + mt * 1024 + lane * 16);
    v16bf a1 = ld16(tA + mt * 1024 + 512 + lane * 16);
    v8f q = zero8(), kk = zero8(), vv = zero8();
    q  = wmma_bf16(a0, ld16(pw_ain + ((h) * 2 + 0) * 512 + lane * 16), q);
    q  = wmma_bf16(a1, ld16(pw_ain + ((h) * 2 + 1) * 512 + lane * 16), q);
    kk = wmma_bf16(a0, ld16(pw_ain + ((4 + h) * 2 + 0) * 512 + lane * 16), kk);
    kk = wmma_bf16(a1, ld16(pw_ain + ((4 + h) * 2 + 1) * 512 + lane * 16), kk);
    vv = wmma_bf16(a0, ld16(pw_ain + ((8 + h) * 2 + 0) * 512 + lane * 16), vv);
    vv = wmma_bf16(a1, ld16(pw_ain + ((8 + h) * 2 + 1) * 512 + lane * 16), vv);
    float qb = ain_b[h * 16 + cc];
    float kb = ain_b[64 + h * 16 + cc];
    float vb = ain_b[128 + h * 16 + cc];
    #pragma unroll
    for(int r = 0; r < 8; r++){
      int ml = cr0 + r;
      int j = mt * 16 + ml;
      aQw[mt * 512 + a_idx(ml, cc)] = f2bf((q[r] + qb) * 0.25f);       // 1/sqrt(16)
      bKw[(j >> 4) * 512 + (j & 15) * 16 + cc] = f2bf(kk[r] + kb);      // K^T tile
      bVw[b_idx(j, cc)] = f2bf(vv[r] + vb);                             // V tile
    }
  }
  wbar();
  #pragma unroll
  for(int mt = 0; mt < 2; mt++){
    v16bf qa = ld16(aQw + mt * 512 + lane * 16);
    #pragma unroll
    for(int ntj = 0; ntj < 2; ntj++){
      v8f s = zero8();
      s = wmma_bf16(qa, ld16(bKw + ntj * 512 + lane * 16), s);
      #pragma unroll
      for(int r = 0; r < 8; r++)
        sbw[(mt * 16 + cr0 + r) * 32 + ntj * 16 + cc] = s[r];
    }
  }
  wbar();
  { // softmax over 24 keys; lane = query row
    int i = lane;
    float mx = -1e30f;
    for(int j = 0; j < 24; j++){ float v = sbw[i * 32 + j]; mx = v > mx ? v : mx; }
    float es = 0.f;
    for(int j = 0; j < 24; j++){ float e = __expf(sbw[i * 32 + j] - mx); sbw[i * 32 + j] = e; es += e; }
    float inv = 1.f / es;
    for(int j = 0; j < 32; j += 2){
      float w0 = (j < 24) ? sbw[i * 32 + j] * inv : 0.f;
      float w1 = (j + 1 < 24) ? sbw[i * 32 + j + 1] * inv : 0.f;
      st_pair(aAWw + (i >> 4) * 512 + a_idx(i & 15, j), w0, w1);
    }
  }
  wbar();
  #pragma unroll
  for(int mt = 0; mt < 2; mt++){
    v16bf aa = ld16(aAWw + mt * 512 + lane * 16);
    v8f ao = zero8();
    ao = wmma_bf16(aa, ld16(bVw + lane * 16), ao);
    #pragma unroll
    for(int r = 0; r < 8; r++)
      aAO[mt * 1024 + a_idx(cr0 + r, h * 16 + cc)] = f2bf(ao[r]);  // concat heads
  }
  __syncthreads();
  #pragma unroll
  for(int mt = 0; mt < 2; mt++){
    v16bf c0 = ld16(aAO + mt * 1024 + lane * 16);
    v16bf c1 = ld16(aAO + mt * 1024 + 512 + lane * 16);
    v8f p = zero8();
    p = wmma_bf16(c0, ld16(pw_aout + (wave * 2 + 0) * 512 + lane * 16), p);
    p = wmma_bf16(c1, ld16(pw_aout + (wave * 2 + 1) * 512 + lane * 16), p);
    int n = wave * 16 + cc;
    float pb = aout_b[n];
    #pragma unroll
    for(int r = 0; r < 8; r++){
      int i = mt * 16 + cr0 + r;
      if(i < 24) attn_out[(r0 + i) * 64 + n] = p[r] + pb;
    }
  }
}

// ---------------------------------------------------------------------------
// K5: feed-forward GRN (identity skip) + LN. 768 WGs x 32 threads.
// ---------------------------------------------------------------------------
__global__ void k5_ffgrn(const float* __restrict__ enr,
    const u16* __restrict__ pw_ff1, const float* __restrict__ ff_b1,
    const u16* __restrict__ pw_ff2, const float* __restrict__ ff_b2,
    const u16* __restrict__ pw_g1, const float* __restrict__ g1b,
    const u16* __restrict__ pw_g2, const float* __restrict__ g2b,
    const float* __restrict__ lng, const float* __restrict__ lnb,
    float* __restrict__ outp)
{
  extern __shared__ char smem[];
  u16* ab = (u16*)smem;               // 1024 halves (K=64)
  u16* fb = (u16*)(smem + 2048);      // 2048 halves (K=128)
  u16* gb = (u16*)(smem + 6144);      // 1024 halves
  float* scr = (float*)(smem + 8192); // 1024 floats
  float* st = (float*)(smem + 12288); // 32 floats
  int lane = threadIdx.x;
  int cr0 = ((lane >> 4) << 3), cc = lane & 15;
  long r0 = (long)blockIdx.x * 16;
  for(int i = lane; i < 512; i += 32){
    int m = i >> 5, d = (i & 31) * 2;
    const float* ip = enr + (r0 + m) * 64 + d;
    st_pair(ab + a_idx(m, d), ip[0], ip[1]);
  }
  wbar();
  v16bf a0 = ld16(ab + lane * 16), a1 = ld16(ab + 512 + lane * 16);
  #pragma unroll
  for(int nt = 0; nt < 8; nt++){
    v8f acc = zero8();
    acc = wmma_bf16(a0, ld16(pw_ff1 + (nt * 2 + 0) * 512 + lane * 16), acc);
    acc = wmma_bf16(a1, ld16(pw_ff1 + (nt * 2 + 1) * 512 + lane * 16), acc);
    int n = nt * 16 + cc;
    float bb = ff_b1[n];
    #pragma unroll
    for(int r = 0; r < 8; r++) fb[a_idx(cr0 + r, n)] = f2bf(eluf(acc[r] + bb));
  }
  wbar();
  v16bf f0 = ld16(fb + lane * 16), f1 = ld16(fb + 512 + lane * 16);
  v16bf f2 = ld16(fb + 1024 + lane * 16), f3 = ld16(fb + 1536 + lane * 16);
  #pragma unroll
  for(int nt = 0; nt < 4; nt++){
    v8f acc = zero8();
    acc = wmma_bf16(f0, ld16(pw_ff2 + (nt * 4 + 0) * 512 + lane * 16), acc);
    acc = wmma_bf16(f1, ld16(pw_ff2 + (nt * 4 + 1) * 512 + lane * 16), acc);
    acc = wmma_bf16(f2, ld16(pw_ff2 + (nt * 4 + 2) * 512 + lane * 16), acc);
    acc = wmma_bf16(f3, ld16(pw_ff2 + (nt * 4 + 3) * 512 + lane * 16), acc);
    int n = nt * 16 + cc;
    float bb = ff_b2[n];
    #pragma unroll
    for(int r = 0; r < 8; r++) gb[a_idx(cr0 + r, n)] = f2bf(acc[r] + bb);
  }
  wbar();
  v16bf g0 = ld16(gb + lane * 16), g1x = ld16(gb + 512 + lane * 16);
  #pragma unroll
  for(int nt = 0; nt < 4; nt++){
    v8f c1 = zero8(), c2 = zero8();
    c1 = wmma_bf16(g0,  ld16(pw_g1 + (nt * 2 + 0) * 512 + lane * 16), c1);
    c1 = wmma_bf16(g1x, ld16(pw_g1 + (nt * 2 + 1) * 512 + lane * 16), c1);
    c2 = wmma_bf16(g0,  ld16(pw_g2 + (nt * 2 + 0) * 512 + lane * 16), c2);
    c2 = wmma_bf16(g1x, ld16(pw_g2 + (nt * 2 + 1) * 512 + lane * 16), c2);
    int n = nt * 16 + cc;
    float b1v = g1b[n], b2v = g2b[n];
    #pragma unroll
    for(int r = 0; r < 8; r++){
      int m = cr0 + r;
      scr[m * 64 + n] = sigm(c1[r] + b1v) * (c2[r] + b2v) + enr[(r0 + m) * 64 + n];
    }
  }
  wbar();
  if(lane < 16){
    float s = 0.f, ss = 0.f;
    for(int d = 0; d < 64; d++){ float v = scr[lane * 64 + d]; s += v; ss += v * v; }
    float mean = s * (1.f / 64.f);
    st[lane * 2] = mean;
    st[lane * 2 + 1] = rsqrtf(ss * (1.f / 64.f) - mean * mean + 1e-5f);
  }
  wbar();
  for(int i = lane; i < 1024; i += 32){
    int m = i >> 6, d = i & 63;
    outp[(r0 + m) * 64 + d] = (scr[m * 64 + d] - st[m * 2]) * st[m * 2 + 1] * lng[d] + lnb[d];
  }
}

// ---------------------------------------------------------------------------
// K6: mean over T, fc1 + relu, fc2 -> [512,6]. 32 WGs x 32 threads.
// ---------------------------------------------------------------------------
__global__ void k6_head(const float* __restrict__ ff,
    const u16* __restrict__ pw_fc1, const float* __restrict__ fc1_b,
    const u16* __restrict__ pw_fc2, const float* __restrict__ fc2_b,
    float* __restrict__ outp)
{
  extern __shared__ char smem[];
  u16* ab = (u16*)smem;
  u16* hb = (u16*)(smem + 2048);
  int lane = threadIdx.x;
  int cr0 = ((lane >> 4) << 3), cc = lane & 15;
  long b0 = (long)blockIdx.x * 16;
  for(int i = lane; i < 512; i += 32){
    int m = i >> 5, d = (i & 31) * 2;
    float s0 = 0.f, s1 = 0.f;
    for(int t = 0; t < 24; t++){
      const float* fp = ff + ((b0 + m) * 24 + t) * 64 + d;
      s0 += fp[0]; s1 += fp[1];
    }
    st_pair(ab + a_idx(m, d), s0 * (1.f / 24.f), s1 * (1.f / 24.f));
  }
  wbar();
  v16bf a0 = ld16(ab + lane * 16), a1 = ld16(ab + 512 + lane * 16);
  #pragma unroll
  for(int nt = 0; nt < 4; nt++){
    v8f acc = zero8();
    acc = wmma_bf16(a0, ld16(pw_fc1 + (nt * 2 + 0) * 512 + lane * 16), acc);
    acc = wmma_bf16(a1, ld16(pw_fc1 + (nt * 2 + 1) * 512 + lane * 16), acc);
    int n = nt * 16 + cc;
    float bb = fc1_b[n];
    #pragma unroll
    for(int r = 0; r < 8; r++) hb[a_idx(cr0 + r, n)] = f2bf(fmaxf(acc[r] + bb, 0.f));
  }
  wbar();
  v16bf h0 = ld16(hb + lane * 16), h1 = ld16(hb + 512 + lane * 16);
  v8f acc = zero8();
  acc = wmma_bf16(h0, ld16(pw_fc2 + 0 * 512 + lane * 16), acc);
  acc = wmma_bf16(h1, ld16(pw_fc2 + 1 * 512 + lane * 16), acc);
  if(cc < 6){
    float bb = fc2_b[cc];
    #pragma unroll
    for(int r = 0; r < 8; r++){
      int m = cr0 + r;
      outp[(b0 + m) * 6 + cc] = acc[r] + bb;
    }
  }
}

// ---------------------------------------------------------------------------
extern "C" void kernel_launch(void* const* d_in, const int* in_sizes, int n_in,
                              void* d_out, int out_size, void* d_ws, size_t ws_size,
                              hipStream_t stream){
  (void)in_sizes; (void)n_in; (void)out_size; (void)ws_size;
  char* ws = (char*)d_ws;
  size_t off = 0;
  auto take = [&](size_t bytes) -> char* {
    char* p = ws + off;
    off = (off + bytes + 255) & ~(size_t)255;
    return p;
  };
  // packed bf16 weight tiles. FW2|FG1|FG2 and WW1|WSK are contiguous blocks so
  // kernels can use a single base pointer + constant immediate offsets.
  u16* P_FW2 = (u16*)take((size_t)3 * 81 * 4096 * 2);     // 3 x 331776 halves
  u16* P_FG1 = P_FW2 + FG1_OFF;
  u16* P_FG2 = P_FW2 + FG2_OFF;
  u16* P_WW1 = (u16*)take((size_t)(331776 + 497664) * 2); // WW1 | WSK
  u16* P_WSK = P_WW1 + WSK_OFF;
  u16* P_WW2 = (u16*)take((size_t)2 * 6 * 512 * 2);
  u16* P_WG1 = (u16*)take((size_t)3 * 6 * 512 * 2);
  u16* P_WG2 = (u16*)take((size_t)3 * 6 * 512 * 2);
  u16* P_WIH = (u16*)take((size_t)2 * 16 * 512 * 2);
  u16* P_WHH = (u16*)take((size_t)2 * 16 * 512 * 2);
  u16* P_PL1 = (u16*)take((size_t)2 * 4 * 512 * 2);
  u16* P_PL2 = (u16*)take((size_t)2 * 4 * 512 * 2);
  u16* P_AIN = (u16*)take((size_t)2 * 12 * 512 * 2);
  u16* P_AOU = (u16*)take((size_t)2 * 4 * 512 * 2);
  u16* P_PA1 = (u16*)take((size_t)2 * 4 * 512 * 2);
  u16* P_PA2 = (u16*)take((size_t)2 * 4 * 512 * 2);
  u16* P_FF1 = (u16*)take((size_t)2 * 8 * 512 * 2);
  u16* P_FF2 = (u16*)take((size_t)4 * 4 * 512 * 2);
  u16* P_FG1b = (u16*)take((size_t)2 * 4 * 512 * 2);
  u16* P_FG2b = (u16*)take((size_t)2 * 4 * 512 * 2);
  u16* P_FC1 = (u16*)take((size_t)2 * 4 * 512 * 2);
  u16* P_FC2 = (u16*)take((size_t)2 * 1 * 512 * 2);
  // activations
  float* SEL = (float*)take((size_t)12288 * 64 * 4);
  float* LSO = (float*)take((size_t)12288 * 64 * 4);
  float* TMP = (float*)take((size_t)12288 * 64 * 4);
  float* AOT = (float*)take((size_t)12288 * 64 * 4);
  float* ENR = (float*)take((size_t)12288 * 64 * 4);
  float* FFO = (float*)take((size_t)12288 * 64 * 4);

  auto pack = [&](const void* src, u16* dst, int K, int N, int KT, int NT,
                  int sK, int sN, int batch, long sBS, long dBS){
    int halves = KT * NT * 512;
    dim3 grid((halves + 255) / 256, batch);
    pack_b<<<grid, dim3(256), 0, stream>>>((const float*)src, dst, K, N, KT, NT, sK, sN, sBS, dBS);
  };
  pack(d_in[3],  P_FW2, 64, 64,  2, 4,  64, 1, 81, 4096, 4096);
  pack(d_in[5],  P_FG1, 64, 64,  2, 4,  64, 1, 81, 4096, 4096);
  pack(d_in[7],  P_FG2, 64, 64,  2, 4,  64, 1, 81, 4096, 4096);
  pack(d_in[13], P_WW1, 5184, 64, 162, 4, 64, 1, 1, 0, 0);
  pack(d_in[21], P_WSK, 5184, 81, 162, 6, 81, 1, 1, 0, 0);
  pack(d_in[15], P_WW2, 64, 81,  2, 6,  81, 1, 1, 0, 0);
  pack(d_in[17], P_WG1, 81, 81,  3, 6,  81, 1, 1, 0, 0);
  pack(d_in[19], P_WG2, 81, 81,  3, 6,  81, 1, 1, 0, 0);
  pack(d_in[25], P_WIH, 64, 256, 2, 16, 1, 64, 1, 0, 0);   // transposed (x @ W^T)
  pack(d_in[26], P_WHH, 64, 256, 2, 16, 1, 64, 1, 0, 0);   // transposed
  pack(d_in[29], P_PL1, 64, 64,  2, 4,  64, 1, 1, 0, 0);
  pack(d_in[31], P_PL2, 64, 64,  2, 4,  64, 1, 1, 0, 0);
  pack(d_in[35], P_AIN, 64, 192, 2, 12, 192, 1, 1, 0, 0);
  pack(d_in[37], P_AOU, 64, 64,  2, 4,  64, 1, 1, 0, 0);
  pack(d_in[39], P_PA1, 64, 64,  2, 4,  64, 1, 1, 0, 0);
  pack(d_in[41], P_PA2, 64, 64,  2, 4,  64, 1, 1, 0, 0);
  pack(d_in[45], P_FF1, 64, 128, 2, 8,  128, 1, 1, 0, 0);
  pack(d_in[47], P_FF2, 128, 64, 4, 4,  64, 1, 1, 0, 0);
  pack(d_in[49], P_FG1b, 64, 64, 2, 4,  64, 1, 1, 0, 0);
  pack(d_in[51], P_FG2b, 64, 64, 2, 4,  64, 1, 1, 0, 0);
  pack(d_in[55], P_FC1, 64, 64,  2, 4,  64, 1, 1, 0, 0);
  pack(d_in[57], P_FC2, 64, 6,   2, 1,  6, 1, 1, 0, 0);

  k1_vsn<<<768, 128, K1_LDS, stream>>>(
      (const float*)d_in[0], (const float*)d_in[1], (const float*)d_in[2],
      (const float*)d_in[4], (const float*)d_in[6], (const float*)d_in[8],
      (const float*)d_in[9], (const float*)d_in[10], (const float*)d_in[11],
      (const float*)d_in[12],
      P_FW2, P_WW1,
      (const float*)d_in[14], P_WW2, (const float*)d_in[16],
      P_WG1, (const float*)d_in[18], P_WG2, (const float*)d_in[20],
      (const float*)d_in[22], (const float*)d_in[23], (const float*)d_in[24],
      SEL);

  k2_lstm<<<32, 128, 90112, stream>>>(SEL, P_WIH, P_WHH,
      (const float*)d_in[27], (const float*)d_in[28], LSO);

  k_glu_ln<<<768, 32, 6272, stream>>>(LSO, SEL,
      P_PL1, (const float*)d_in[30], P_PL2, (const float*)d_in[32],
      (const float*)d_in[33], (const float*)d_in[34], TMP);

  k4_attn<<<512, 128, 53248, stream>>>(TMP, P_AIN, (const float*)d_in[36],
      P_AOU, (const float*)d_in[38], AOT);

  k_glu_ln<<<768, 32, 6272, stream>>>(AOT, TMP,
      P_PA1, (const float*)d_in[40], P_PA2, (const float*)d_in[42],
      (const float*)d_in[43], (const float*)d_in[44], ENR);

  k5_ffgrn<<<768, 32, 12416, stream>>>(ENR,
      P_FF1, (const float*)d_in[46], P_FF2, (const float*)d_in[48],
      P_FG1b, (const float*)d_in[50], P_FG2b, (const float*)d_in[52],
      (const float*)d_in[53], (const float*)d_in[54], FFO);

  k6_head<<<32, 32, 4096, stream>>>(FFO,
      P_FC1, (const float*)d_in[56], P_FC2, (const float*)d_in[58],
      (float*)d_out);
}
